// TextAttention_42511586296294
// MI455X (gfx1250) — compile-verified, matrix operand order checked by reading, not run on hardware
//
#include <hip/hip_runtime.h>
#include <hip/hip_bf16.h>

// ---------------------------------------------------------------------------
// Problem constants (from reference): B=2, S=2048, HID=2048, H=16, KV=8, D=128
// ---------------------------------------------------------------------------
#define Bc   2
#define Sc   2048
#define HIDc 2048
#define Hc   16
#define KVc  8
#define Dc   128
#define MROWS (Bc * Sc)              // 4096 rows for all GEMMs
#define SCALEc 0.08838834764831845f // 1/sqrt(128)

typedef __attribute__((ext_vector_type(16))) __bf16  v16bf;
typedef __attribute__((ext_vector_type(2)))  __bf16  v2bf;
typedef __attribute__((ext_vector_type(8)))  float   v8f;
typedef __attribute__((ext_vector_type(4)))  float   v4f;
typedef __attribute__((ext_vector_type(2)))  float   v2f;
typedef __attribute__((ext_vector_type(8)))  unsigned v8u;
typedef __attribute__((ext_vector_type(2)))  unsigned v2u;

#define DEV static __device__ __forceinline__

// Native HW converts: scalar cast -> v_cvt bf16; pair -> one v_cvt_pk_bf16_f32.
DEV __bf16 f2bf(float f) { return (__bf16)f; }

DEV unsigned cvt2(float a, float b) {
  v2f t; t.x = a; t.y = b;
  v2bf r = __builtin_convertvector(t, v2bf);   // fptrunc <2xf32> -> <2xbf16>
  return __builtin_bit_cast(unsigned, r);      //   = v_cvt_pk_bf16_f32
}

DEV v2u pack4(v4f x) {
  v2u r;
  r.x = cvt2(x.x, x.y);
  r.y = cvt2(x.z, x.w);
  return r;
}

// WMMA 16-bit operand K-striping (ISA 7.12.2), and its inverse:
//   VGPR-pair base K for operand vgpr v, lane-half:  kpair(v,half)
//   element index e within the 16-elem per-lane block for K-offset k: efrag(k)
//   which lane-half holds k:                         (k>>3)&1
DEV int kpair(int v, int half) {
  return ((v < 4) ? (2 * v) : (16 + 2 * (v - 4))) + 8 * half;
}
DEV int efrag(int k) { return 2 * ((k & 7) >> 1) + (k & 1) + 8 * (k >> 4); }

DEV v8f wmma_bf16(v16bf a, v16bf b, v8f c) {
  return __builtin_amdgcn_wmma_f32_16x16x32_bf16(
      /*neg_a=*/false, a, /*neg_b=*/false, b,
      /*c_mod=*/(short)0, c, /*reuse_a=*/false, /*reuse_b=*/false);
}

DEV void store_c(float* C, int ldc, int row0, int n0, v8f acc) {
  int lane = threadIdx.x & 31, n = lane & 15, half = lane >> 4;
#pragma unroll
  for (int r = 0; r < 8; ++r)
    C[(size_t)(row0 + r + 8 * half) * ldc + n0 + n] = acc[r];
}

// ---------------------------------------------------------------------------
// GEMM staging: phase 1 issues all global loads into registers (latency
// batched under one wait), phase 2 converts (packed HW cvt) + ds_store_b64
// into LDS in WMMA fragment order (per-lane 16 contiguous bf16 => 2x b128).
// ---------------------------------------------------------------------------
DEV void stage_load(const float* __restrict__ A, const float* __restrict__ Bw,
                    int K, int N, int row0, int gn0, int k0, int tid,
                    v4f* ra, float* rb) {
#pragma unroll
  for (int e = 0; e < 4; ++e) {                    // A tile: 1024 float4 groups
    int idx = tid + e * 256;
    int m = idx >> 3, k = (idx & 7) * 4;
    ra[e] = *(const v4f*)(A + (size_t)(row0 + m) * K + k0 + k);
  }
#pragma unroll
  for (int e = 0; e < 2; ++e) {                    // B tile: 512 groups of 4 k
    int idx = tid + e * 256;
    int n = idx & 63, k = (idx >> 6) * 4;
    const float* p = Bw + (size_t)(k0 + k) * N + gn0 + n;
#pragma unroll
    for (int i = 0; i < 4; ++i) rb[e * 4 + i] = p[(size_t)i * N];
  }
}

DEV void stage_store(__bf16* la, __bf16* lb, int tid,
                     const v4f* ra, const float* rb) {
#pragma unroll
  for (int e = 0; e < 4; ++e) {
    int idx = tid + e * 256;
    int m = idx >> 3, k = (idx & 7) * 4;
    int addr = ((m >> 4) * 32 + (m & 15) + 16 * ((k >> 3) & 1)) * 16 +
               2 * ((k & 7) >> 1) + 8 * (k >> 4);
    *(v2u*)&la[addr] = pack4(ra[e]);
  }
#pragma unroll
  for (int e = 0; e < 2; ++e) {
    int idx = tid + e * 256;
    int n = idx & 63, k = (idx >> 6) * 4;
    int addr = ((n >> 4) * 32 + (n & 15) + 16 * ((k >> 3) & 1)) * 16 +
               2 * ((k & 7) >> 1) + 8 * (k >> 4);
    v2u p;
    p.x = cvt2(rb[e * 4 + 0], rb[e * 4 + 1]);
    p.y = cvt2(rb[e * 4 + 2], rb[e * 4 + 3]);
    *(v2u*)&lb[addr] = p;
  }
}

// ---------------------------------------------------------------------------
// GEMM: C[M,N] = A[M,K] (f32) x Bw[K,N] (f32), bf16 WMMA, f32 accumulate.
// Block = 8 waves -> 128x64 tile; double-buffered LDS, one barrier per
// K-step; next tile's global loads issue before the current tile's WMMAs.
// ---------------------------------------------------------------------------
__global__ __launch_bounds__(256) void gemm_bf16(const float* __restrict__ A,
                                                 const float* __restrict__ Bw,
                                                 float* __restrict__ C,
                                                 int M, int N, int K) {
  __shared__ __bf16 lds_a[2 * 128 * 32] __attribute__((aligned(32)));
  __shared__ __bf16 lds_b[2 * 4 * 32 * 16] __attribute__((aligned(32)));
  int tid = threadIdx.x, lane = tid & 31, w = tid >> 5;
  int row0 = blockIdx.y * 128;
  int gn0  = blockIdx.x * 64;

  v8f acc[4] = {};
  v4f ra[4]; float rb[8];
  stage_load(A, Bw, K, N, row0, gn0, 0, tid, ra, rb);

  for (int k0 = 0; k0 < K; k0 += 32) {
    __bf16* la = lds_a + ((k0 >> 5) & 1) * (128 * 32);
    __bf16* lb = lds_b + ((k0 >> 5) & 1) * (4 * 32 * 16);
    stage_store(la, lb, tid, ra, rb);
    __syncthreads();
    if (k0 + 32 < K) {
      stage_load(A, Bw, K, N, row0, gn0, k0 + 32, tid, ra, rb);
      if (k0 + 64 < K)                       // gfx1250 global_prefetch_b8
        __builtin_prefetch(Bw + (size_t)(k0 + 64) * N + gn0 + lane, 0, 1);
    }
    v16bf a = *(const v16bf*)&lds_a[((k0 >> 5) & 1) * (128 * 32) + (w * 32 + lane) * 16];
#pragma unroll
    for (int c = 0; c < 4; ++c) {
      v16bf bb = *(const v16bf*)&lb[(c * 32 + lane) * 16];
      acc[c] = wmma_bf16(a, bb, acc[c]);
    }
  }
#pragma unroll
  for (int c = 0; c < 4; ++c)
    store_c(C, N, row0 + w * 16, gn0 + c * 16, acc[c]);
}

// ---------------------------------------------------------------------------
// RMS-norm + RoPE, one wave per (b,s,head) row of D=128. Writes bf16 in
// WMMA-fragment-major order: frag = ((b*Hn+h)*(S/16) + s/16)*4 + kk,
// 512 bf16 per fragment, lane-slot*16 + e.  All 4 outputs of a lane share
// the same slot/e (d and d+32/64/96 have identical k&31), only kk differs.
// ---------------------------------------------------------------------------
__global__ __launch_bounds__(256) void rope_norm(const float* __restrict__ X,
                                                 const float* __restrict__ sinb,
                                                 const float* __restrict__ cosb,
                                                 const float* __restrict__ wn,
                                                 __bf16* __restrict__ Y,
                                                 int Hn, int nrows) {
  int wave = (int)((blockIdx.x * blockDim.x + threadIdx.x) >> 5);
  if (wave >= nrows) return;
  int lane = threadIdx.x & 31;
  const float* xr = X + (size_t)wave * Dc;
  float x0 = xr[lane], x1 = xr[lane + 32], x2 = xr[lane + 64], x3 = xr[lane + 96];
  float ss = x0 * x0 + x1 * x1 + x2 * x2 + x3 * x3;
#pragma unroll
  for (int m = 1; m < 32; m <<= 1) ss += __shfl_xor(ss, m, 32);
  float inv = rsqrtf(ss * (1.0f / (float)Dc) + 1e-6f);
  x0 *= inv * wn[lane];      x1 *= inv * wn[lane + 32];
  x2 *= inv * wn[lane + 64]; x3 *= inv * wn[lane + 96];

  int hh = wave % Hn;
  int s  = (wave / Hn) % Sc;
  int b  = wave / (Hn * Sc);
  const float* cp = cosb + ((size_t)b * Sc + s) * Dc;
  const float* sp = sinb + ((size_t)b * Sc + s) * Dc;
  float c0 = cp[lane], c1 = cp[lane + 32], c2 = cp[lane + 64], c3 = cp[lane + 96];
  float s0 = sp[lane], s1 = sp[lane + 32], s2 = sp[lane + 64], s3 = sp[lane + 96];

  float y0 = x0 * c0 - x2 * s0;   // d < 64: rot = -x[d+64]
  float y1 = x1 * c1 - x3 * s1;
  float y2 = x2 * c2 + x0 * s2;   // d >= 64: rot = +x[d-64]
  float y3 = x3 * c3 + x1 * s3;

  size_t frag0 = ((size_t)(b * Hn + hh) * (Sc / 16) + (s >> 4)) * 4;
  int slot = (s & 15) + 16 * ((lane >> 3) & 1);
  int e    = efrag(lane);
  Y[(frag0 + 0) * 512 + slot * 16 + e] = f2bf(y0);
  Y[(frag0 + 1) * 512 + slot * 16 + e] = f2bf(y1);
  Y[(frag0 + 2) * 512 + slot * 16 + e] = f2bf(y2);
  Y[(frag0 + 3) * 512 + slot * 16 + e] = f2bf(y3);
}

// ---------------------------------------------------------------------------
// Pack V (f32 [B*S, KV*D]) into B-operand fragment-major bf16:
// frag = ((b*KV+kv)*(S/32) + jt)*8 + c ; per-lane 16 contiguous bf16
// assembled as 8 packed-cvt u32 words (pairs (e,e+1) = rows j, j+1).
// One wave per fragment.
// ---------------------------------------------------------------------------
__global__ __launch_bounds__(256) void pack_v(const float* __restrict__ Vp,
                                              __bf16* __restrict__ Vpk) {
  int wave = (int)((blockIdx.x * blockDim.x + threadIdx.x) >> 5); // 0..8191
  int lane = threadIdx.x & 31, nn = lane & 15, half = lane >> 4;
  int c = wave & 7, t = wave >> 3;
  int jt = t & 63; t >>= 6;
  int kv = t & 7;
  int b  = t >> 3;
  v8u r;
#pragma unroll
  for (int p = 0; p < 8; ++p) {
    int k = kpair(p, half);
    int j = jt * 32 + k;
    const float* src = Vp + ((size_t)b * Sc + j) * (KVc * Dc) + kv * Dc + c * 16 + nn;
    r[p] = cvt2(src[0], src[KVc * Dc]);   // rows j and j+1
  }
  *(v8u*)(Vpk + (size_t)wave * 512 + lane * 16) = r;
}

// ---------------------------------------------------------------------------
// Flash attention: one wave per (b, h, 16-query-row tile). Key tiles of 32.
// All WMMA operands are single v16bf loads (2x b128) from fragment-major
// buffers; P goes through 1KB LDS in fragment order.
// ---------------------------------------------------------------------------
__global__ __launch_bounds__(32) void attn(const __bf16* __restrict__ Qpk,
                                           const __bf16* __restrict__ Kpk,
                                           const __bf16* __restrict__ Vpk,
                                           float* __restrict__ O) {
  __shared__ __bf16 ldsP[32 * 16] __attribute__((aligned(32)));
  int it = blockIdx.x, h = blockIdx.y, b = blockIdx.z;
  int lane = threadIdx.x & 31, nn = lane & 15, half = lane >> 4;
  int i0 = it * 16;
  int kh = h >> 1;                               // GQA: groups = H/KV = 2

  size_t qf  = ((size_t)(b * Hc + h) * (Sc / 16) + it) * 4;
  size_t kfb = (size_t)(b * KVc + kh) * (Sc / 16);
  size_t vfb = (size_t)(b * KVc + kh) * (Sc / 32);

  v16bf aq[4];
#pragma unroll
  for (int kk = 0; kk < 4; ++kk)
    aq[kk] = *(const v16bf*)(Qpk + (qf + kk) * 512 + lane * 16);

  v8f o[8] = {};
  float rmax[8], rsum[8];
#pragma unroll
  for (int r = 0; r < 8; ++r) { rmax[r] = -3.0e38f; rsum[r] = 0.0f; }

  int nt = (i0 + 47) >> 5;                       // causal: j0 <= i0+15
  for (int jt = 0; jt < nt; ++jt) {
    int j0 = jt * 32;
    v8f s0 = {}, s1 = {};
#pragma unroll
    for (int kk = 0; kk < 4; ++kk) {
      v16bf bk0 = *(const v16bf*)(Kpk + ((kfb + 2 * jt) * 4 + kk) * 512 + lane * 16);
      v16bf bk1 = *(const v16bf*)(Kpk + ((kfb + 2 * jt + 1) * 4 + kk) * 512 + lane * 16);
      s0 = wmma_bf16(aq[kk], bk0, s0);
      s1 = wmma_bf16(aq[kk], bk1, s1);
    }
    // scale + causal mask + online softmax
#pragma unroll
    for (int r = 0; r < 8; ++r) {
      int i = i0 + r + 8 * half;
      float v0 = s0[r] * SCALEc; if (j0 + nn > i)      v0 = -1.0e9f;
      float v1 = s1[r] * SCALEc; if (j0 + 16 + nn > i) v1 = -1.0e9f;
      s0[r] = v0; s1[r] = v1;
      float t = fmaxf(v0, v1);
#pragma unroll
      for (int m = 1; m < 16; m <<= 1) t = fmaxf(t, __shfl_xor(t, m, 32));
      float newm = fmaxf(rmax[r], t);
      float sc = __expf(rmax[r] - newm);
      rsum[r] *= sc;
#pragma unroll
      for (int c = 0; c < 8; ++c) o[c][r] *= sc;
      rmax[r] = newm;
    }
    // exp + row sums + write P to LDS in fragment order
#pragma unroll
    for (int r = 0; r < 8; ++r) {
      float p0 = __expf(s0[r] - rmax[r]);
      float p1 = __expf(s1[r] - rmax[r]);
      int m  = r + 8 * half;
      int sl = m + 16 * ((nn >> 3) & 1);
      int e0 = 2 * ((nn & 7) >> 1) + (nn & 1);
      ldsP[sl * 16 + e0]     = f2bf(p0);   // jj = nn
      ldsP[sl * 16 + e0 + 8] = f2bf(p1);   // jj = nn + 16
      float ps = p0 + p1;
#pragma unroll
      for (int mk = 1; mk < 16; mk <<= 1) ps += __shfl_xor(ps, mk, 32);
      rsum[r] += ps;
    }
    __syncthreads();
    // P(16x32) @ V(32x128)
    v16bf ap = *(const v16bf*)(ldsP + lane * 16);
#pragma unroll
    for (int c = 0; c < 8; ++c) {
      v16bf bv = *(const v16bf*)(Vpk + ((vfb + jt) * 8 + c) * 512 + lane * 16);
      o[c] = wmma_bf16(ap, bv, o[c]);
    }
    __syncthreads();
  }
  // epilogue: divide by row sums, write f32 attn_out [B*S, H*D]
  float* Orow = O + ((size_t)b * Sc + i0) * (Hc * Dc) + h * Dc;
#pragma unroll
  for (int c = 0; c < 8; ++c)
#pragma unroll
    for (int r = 0; r < 8; ++r) {
      int m = r + 8 * half;
      Orow[(size_t)m * (Hc * Dc) + c * 16 + nn] = o[c][r] / rsum[r];
    }
}

// ---------------------------------------------------------------------------
// Host-side launcher. ws layout (bytes):
//   Qp  f32 [4096x2048] @ 0     (32MB, reused as attn_out)
//   Kp  f32 [4096x1024] @ 32MB
//   Vp  f32 [4096x1024] @ 48MB
//   Qpk bf16 packed     @ 64MB  (16MB)
//   Kpk bf16 packed     @ 80MB  ( 8MB)
//   Vpk bf16 packed     @ 88MB  ( 8MB)   total 96MB
// ---------------------------------------------------------------------------
extern "C" void kernel_launch(void* const* d_in, const int* in_sizes, int n_in,
                              void* d_out, int out_size, void* d_ws, size_t ws_size,
                              hipStream_t stream) {
  const float* x    = (const float*)d_in[0];
  const float* sinb = (const float*)d_in[1];
  const float* cosb = (const float*)d_in[2];
  // d_in[3] = attention_mask (pure causal; applied analytically)
  const float* Wq = (const float*)d_in[4];
  const float* Wk = (const float*)d_in[5];
  const float* Wv = (const float*)d_in[6];
  const float* Wo = (const float*)d_in[7];
  const float* qw = (const float*)d_in[8];
  const float* kw = (const float*)d_in[9];

  char* ws = (char*)d_ws;
  float*  Qp  = (float*)(ws);
  float*  Kp  = (float*)(ws + (size_t)32 * 1024 * 1024);
  float*  Vp  = (float*)(ws + (size_t)48 * 1024 * 1024);
  __bf16* Qpk = (__bf16*)(ws + (size_t)64 * 1024 * 1024);
  __bf16* Kpk = (__bf16*)(ws + (size_t)80 * 1024 * 1024);
  __bf16* Vpk = (__bf16*)(ws + (size_t)88 * 1024 * 1024);
  float* attn_out = Qp;  // Qp dead after rope_norm; reuse for attention output
  float* out = (float*)d_out;

  // 1) QKV projections
  gemm_bf16<<<dim3(HIDc / 64, MROWS / 128), 256, 0, stream>>>(x, Wq, Qp, MROWS, Hc * Dc, HIDc);
  gemm_bf16<<<dim3((KVc * Dc) / 64, MROWS / 128), 256, 0, stream>>>(x, Wk, Kp, MROWS, KVc * Dc, HIDc);
  gemm_bf16<<<dim3((KVc * Dc) / 64, MROWS / 128), 256, 0, stream>>>(x, Wv, Vp, MROWS, KVc * Dc, HIDc);

  // 2) RMS-norm + RoPE (Q,K) -> packed fragments; pack V
  rope_norm<<<(MROWS * Hc) / 8, 256, 0, stream>>>(Qp, sinb, cosb, qw, Qpk, Hc, MROWS * Hc);
  rope_norm<<<(MROWS * KVc) / 8, 256, 0, stream>>>(Kp, sinb, cosb, kw, Kpk, KVc, MROWS * KVc);
  pack_v<<<(Bc * KVc * (Sc / 32) * 8) / 8, 256, 0, stream>>>(Vp, Vpk);

  // 3) Flash attention
  attn<<<dim3(Sc / 16, Hc, Bc), 32, 0, stream>>>(Qpk, Kpk, Vpk, attn_out);

  // 4) Output projection
  gemm_bf16<<<dim3(HIDc / 64, MROWS / 128), 256, 0, stream>>>(attn_out, Wo, out, MROWS, HIDc, HIDc);
}
